// Adaptive_SFMCell_85736137163011
// MI455X (gfx1250) — matrix operand order repaired and checked by reading, not compiled, not run on hardware
//
#include <hip/hip_runtime.h>
#include <hip/hip_bf16.h>

// Problem constants (from reference): B=16, T=64, I=192, U=192
#define Bb 16
#define Tt 64
#define Ii 192
#define Uu 192
#define G5 (5 * Uu)          // 960
#define KBU (Uu * Bb * Uu)   // 589824, (k,b,u) state layout
#define KT6 (Uu / 32)        // 6  K-tiles per 192 contraction
#define NT12 (Uu / 16)       // 12 N-tiles per 192 columns
#define NT60 (G5 / 16)       // 60 N-tiles for the 5-gate GEMM

typedef __attribute__((ext_vector_type(16))) __bf16 v16bf;
typedef __attribute__((ext_vector_type(8)))  float v8f;

#define WMMA_BF16(a, b, c) \
  __builtin_amdgcn_wmma_f32_16x16x32_bf16(false, (a), false, (b), (short)0, (c), false, false)

__device__ __forceinline__ unsigned short f2bf(float f) {
  unsigned int u = __float_as_uint(f);
  unsigned int r = u + 0x7FFFu + ((u >> 16) & 1u);
  return (unsigned short)(r >> 16);
}

__device__ __forceinline__ float hsig(float v) {
  return fminf(fmaxf(0.2f * v + 0.5f, 0.0f), 1.0f);
}

// ---------------------------------------------------------------------------
// Packed fragment layouts (wave32, v_wmma_f32_16x16x32_bf16):
//  A (16x32 MxK): frag elem e of lane (m=lane&15, hb=lane>>4):
//     e<8 -> K = hb*8 + e ; e>=8 -> K = hb*8 + 16 + (e-8)
//  B (32x16 KxN): frag elem j of lane (n=lane&15): same K order as A (hb = lane>>4)
//  Packed storage: tile-blocked, [tile][lane][16] bf16 = 32 contiguous
//  bytes per lane -> fragment load is a single 32B vector load per lane.
// ---------------------------------------------------------------------------

__global__ void zero_kernel(float* __restrict__ p, int n) {
  for (int i = blockIdx.x * blockDim.x + threadIdx.x; i < n; i += gridDim.x * blockDim.x)
    p[i] = 0.0f;
}

// One-time fp32 -> packed-bf16 B-matrix conversion (nmat matrices of
// KTOT x NTOT, row stride ldb, column offset coloff within each matrix).
__global__ void pack_b_kernel(const float* __restrict__ W, unsigned short* __restrict__ out,
                              int nmat, int matStride, int KT, int NT, int ldb,
                              int coloff, int total) {
  for (int idx = blockIdx.x * blockDim.x + threadIdx.x; idx < total;
       idx += gridDim.x * blockDim.x) {
    int j    = idx & 15;
    int lane = (idx >> 4) & 31;
    int tn   = idx >> 9;                 // (m, kt, nt)
    int perM = KT * NT;
    int m    = tn / perM;
    int tt   = tn - m * perM;
    int kt   = tt / NT;
    int nt   = tt - kt * NT;
    int k    = kt * 32 + (lane >> 4) * 8 + ((j < 8) ? j : j + 8);
    int n    = nt * 16 + (lane & 15);
    out[idx] = f2bf(W[(size_t)m * matStride + (size_t)k * ldb + coloff + n]);
  }
}

// Per-step: pack one 16x192 fp32 matrix (row stride lda) into A-fragment
// layout: [kt][lane][16] bf16, 6*32*16 = 3072 elements.
__global__ void pack_a_kernel(const float* __restrict__ src, int lda,
                              unsigned short* __restrict__ dst) {
  int idx = blockIdx.x * blockDim.x + threadIdx.x;
  if (idx >= KT6 * 512) return;
  int e    = idx & 15;
  int lane = (idx >> 4) & 31;
  int kt   = idx >> 9;
  int m    = lane & 15;
  int hb   = lane >> 4;
  int q    = (hb == 0) ? ((e < 8) ? e : e + 8) : ((e < 8) ? e + 8 : e + 16);
  dst[idx] = f2bf(src[m * lda + kt * 32 + q]);
}

// Stage 1: gates = x_t @ kernel + z_prev @ recurrent + bias; per-gate
// activations; omega_next (raw xw gate) into ping-pong buffer.
// Grid: 60 blocks (one wave per 16-wide column tile).
__global__ void gates_kernel(const unsigned short* __restrict__ xpk,
                             const unsigned short* __restrict__ zpk,
                             const unsigned short* __restrict__ Kpk,
                             const unsigned short* __restrict__ Rpk,
                             const float* __restrict__ bias,
                             float* __restrict__ act, float* __restrict__ omg_next) {
  int nt   = blockIdx.x;              // 0..59
  int lane = threadIdx.x & 31;
  const v16bf* xf = (const v16bf*)xpk;
  const v16bf* zf = (const v16bf*)zpk;
  const v16bf* Kf = (const v16bf*)Kpk;
  const v16bf* Rf = (const v16bf*)Rpk;
  v8f acc = {};
#pragma unroll
  for (int kk = 0; kk < KT6; ++kk) {
    acc = WMMA_BF16(xf[kk * 32 + lane], Kf[(kk * NT60 + nt) * 32 + lane], acc);
    acc = WMMA_BF16(zf[kk * 32 + lane], Rf[(kk * NT60 + nt) * 32 + lane], acc);
  }
  int col = nt * 16 + (lane & 15);
  int g = col / Uu;                   // gate 0..4 (tile never straddles gates)
  float bv = bias[col];
#pragma unroll
  for (int r = 0; r < 8; ++r) {
    int m = r + 8 * (lane >> 4);      // batch row (C/D layout)
    float v = acc[r] + bv;
    float o = (g < 3) ? hsig(v) : ((g == 3) ? tanhf(v) : v);
    act[m * G5 + col] = o;
    if (g == 4) omg_next[m * Uu + (col - 4 * Uu)] = v;
  }
}

// Stage 2: oscillator update over (k,b,u); A written directly in packed
// A-fragment layout so stage 3 does contiguous 32B/lane fragment loads.
__global__ void state_kernel(const float* __restrict__ act, const float* __restrict__ omg_prev,
                             const float* __restrict__ tin, float* __restrict__ Re,
                             float* __restrict__ Im, unsigned short* __restrict__ Apk,
                             int tstep) {
  int idx = blockIdx.x * blockDim.x + threadIdx.x;
  if (idx >= KBU) return;
  int k = idx / (Bb * Uu);
  int r = idx - k * (Bb * Uu);
  int b = r / Uu;
  int u = r - b * Uu;
  const float* ab = act + b * G5;
  float ig  = ab[u] * ab[3 * Uu + u];          // i_g * mod
  float fre = ab[Uu + k];
  float ste = ab[2 * Uu + u];
  float ph  = omg_prev[b * Uu + k] * tin[b * Tt + tstep];
  float sn, cs;
  __sincosf(ph, &sn, &cs);
  float fg = ste * fre;
  float re = fg * Re[idx] + ig * cs;
  float im = fg * Im[idx] + ig * sn;
  Re[idx] = re;
  Im[idx] = im;
  // packed A-fragment address for element (k, m=b, u)
  int q  = u & 31;
  int kt = u >> 5;
  int hb, e;
  if (q < 8)       { hb = 0; e = q; }
  else if (q < 16) { hb = 1; e = q - 8; }
  else if (q < 24) { hb = 0; e = q - 8; }
  else             { hb = 1; e = q - 16; }
  int lane = b + 16 * hb;
  Apk[(((size_t)k * KT6 + kt) * 32 + lane) * 16 + e] = f2bf(sqrtf(re * re + im * im));
}

// Stage 3: per-k block: S1 = A_k@Uo[k] + x_t@fki[k] + z@Vo[k]; S2 = A_k@Wz[k];
// zz[k] = hsig(S1+bo) * tanh(S2+bz). 12 waves/block, one N-tile per wave.
// Inner loop: pure contiguous 32B fragment loads + WMMA.
__global__ void freq_kernel(const unsigned short* __restrict__ Apk,
                            const unsigned short* __restrict__ xpk,
                            const unsigned short* __restrict__ zpk,
                            const unsigned short* __restrict__ Upk,
                            const unsigned short* __restrict__ Vpk,
                            const unsigned short* __restrict__ Wpk,
                            const unsigned short* __restrict__ Fpk,
                            const float* __restrict__ fb, float* __restrict__ zz) {
  int k    = blockIdx.x;
  int nt   = threadIdx.x >> 5;        // 0..11
  int lane = threadIdx.x & 31;
  const v16bf* Af = (const v16bf*)Apk;
  const v16bf* xf = (const v16bf*)xpk;
  const v16bf* zf = (const v16bf*)zpk;
  const v16bf* Uf = (const v16bf*)Upk;
  const v16bf* Vf = (const v16bf*)Vpk;
  const v16bf* Wf = (const v16bf*)Wpk;
  const v16bf* Ff = (const v16bf*)Fpk;
  v8f accU = {}, accW = {};
#pragma unroll
  for (int kk = 0; kk < KT6; ++kk) {
    v16bf aA = Af[((size_t)k * KT6 + kk) * 32 + lane];
    size_t bi = (((size_t)k * KT6 + kk) * NT12 + nt) * 32 + lane;
    accU = WMMA_BF16(aA, Uf[bi], accU);
    accW = WMMA_BF16(aA, Wf[bi], accW);
    accU = WMMA_BF16(zf[kk * 32 + lane], Vf[bi], accU);
    accU = WMMA_BF16(xf[kk * 32 + lane], Ff[bi], accU);
  }
  int v = nt * 16 + (lane & 15);
  float bo = fb[k * Uu + v];
  float bz = fb[(Uu + k) * Uu + v];
#pragma unroll
  for (int r = 0; r < 8; ++r) {
    int m = r + 8 * (lane >> 4);
    float o = hsig(accU[r] + bo);
    zz[(size_t)k * (Bb * Uu) + m * Uu + v] = o * tanhf(accW[r] + bz);
  }
}

// Stage 4: deterministic reduction over k; out[b][t][v] is also the z carry.
__global__ void reduce_kernel(const float* __restrict__ zz, float* __restrict__ out, int tstep) {
  int idx = blockIdx.x * blockDim.x + threadIdx.x;  // b*U + v
  if (idx >= Bb * Uu) return;
  float s = 0.0f;
  for (int k = 0; k < Uu; ++k) s += zz[(size_t)k * (Bb * Uu) + idx];
  int b = idx / Uu;
  int v = idx - b * Uu;
  out[b * (Tt * Uu) + tstep * Uu + v] = s;
}

// ---------------------------------------------------------------------------
extern "C" void kernel_launch(void* const* d_in, const int* in_sizes, int n_in,
                              void* d_out, int out_size, void* d_ws, size_t ws_size,
                              hipStream_t stream) {
  const float* x    = (const float*)d_in[0];  // (B,T,I)
  const float* tin  = (const float*)d_in[1];  // (B,T,1)
  const float* kern = (const float*)d_in[2];  // (I,5U)
  const float* rker = (const float*)d_in[3];  // (U,5U)
  const float* fk   = (const float*)d_in[4];  // (U,U,3U): Uo|Vo|Wz
  const float* fki  = (const float*)d_in[5];  // (U,I,U)
  const float* bias = (const float*)d_in[6];  // (5U,)
  const float* fb   = (const float*)d_in[7];  // (2U,U)
  float* out = (float*)d_out;                 // (B,T,U)

  // ---- workspace layout (~66 MB total) ----
  // fp32 region (zero-initialized part first, contiguous):
  float* Re    = (float*)d_ws;          // KBU
  float* Im    = Re + KBU;              // KBU
  float* omg0  = Im + KBU;              // B*U
  float* omg1  = omg0 + Bb * Uu;        // B*U
  float* zzero = omg1 + Bb * Uu;        // B*U
  float* act   = zzero + Bb * Uu;       // B*5U
  float* zz    = act + Bb * G5;         // KBU
  // packed bf16 region (32B-aligned: preceding float region is 32B-multiple):
  unsigned short* Apk = (unsigned short*)(zz + KBU);   // KBU
  unsigned short* xpk = Apk + KBU;                     // 3072
  unsigned short* zpk = xpk + KT6 * 512;               // 3072
  unsigned short* Kpk = zpk + KT6 * 512;               // 184320
  unsigned short* Rpk = Kpk + Ii * G5;                 // 184320
  unsigned short* Upk = Rpk + Uu * G5;                 // 7077888
  unsigned short* Vpk = Upk + (size_t)Uu * Uu * Uu;
  unsigned short* Wpk = Vpk + (size_t)Uu * Uu * Uu;
  unsigned short* Fpk = Wpk + (size_t)Uu * Uu * Uu;

  // Re-init carry state every call (graph replay reuses ws).
  int nz = 2 * KBU + 3 * Bb * Uu;
  zero_kernel<<<1024, 256, 0, stream>>>(Re, nz);

  // One-time (per launch) weight conversion into packed bf16 fragments.
  int bigTot = Uu * Uu * Uu;            // 7,077,888 per sub-matrix
  pack_b_kernel<<<1024, 256, 0, stream>>>(kern, Kpk, 1, 0, KT6, NT60, G5, 0, Ii * G5);
  pack_b_kernel<<<1024, 256, 0, stream>>>(rker, Rpk, 1, 0, KT6, NT60, G5, 0, Uu * G5);
  pack_b_kernel<<<4096, 256, 0, stream>>>(fk, Upk, Uu, Uu * 3 * Uu, KT6, NT12, 3 * Uu, 0,      bigTot);
  pack_b_kernel<<<4096, 256, 0, stream>>>(fk, Vpk, Uu, Uu * 3 * Uu, KT6, NT12, 3 * Uu, Uu,     bigTot);
  pack_b_kernel<<<4096, 256, 0, stream>>>(fk, Wpk, Uu, Uu * 3 * Uu, KT6, NT12, 3 * Uu, 2 * Uu, bigTot);
  pack_b_kernel<<<4096, 256, 0, stream>>>(fki, Fpk, Uu, Ii * Uu, KT6, NT12, Uu, 0, bigTot);

  for (int t = 0; t < Tt; ++t) {
    const float* zprev = (t == 0) ? zzero : (out + (t - 1) * Uu);
    int zld = (t == 0) ? Uu : (Tt * Uu);
    float* omg_prev = (t & 1) ? omg1 : omg0;
    float* omg_next = (t & 1) ? omg0 : omg1;

    pack_a_kernel<<<12, 256, 0, stream>>>(x + t * Ii, Tt * Ii, xpk);
    pack_a_kernel<<<12, 256, 0, stream>>>(zprev, zld, zpk);
    gates_kernel<<<NT60, 32, 0, stream>>>(xpk, zpk, Kpk, Rpk, bias, act, omg_next);
    state_kernel<<<(KBU + 255) / 256, 256, 0, stream>>>(act, omg_prev, tin, Re, Im, Apk, t);
    freq_kernel<<<Uu, NT12 * 32, 0, stream>>>(Apk, xpk, zpk, Upk, Vpk, Wpk, Fpk, fb, zz);
    reduce_kernel<<<(Bb * Uu + 255) / 256, 256, 0, stream>>>(zz, out, t);
  }
}
